// ParamEncoder_52931176956032
// MI455X (gfx1250) — matrix operand order repaired and checked by reading, not compiled, not run on hardware
//
#include <hip/hip_runtime.h>

typedef __attribute__((ext_vector_type(8)))  float  v8f;
typedef __attribute__((ext_vector_type(8)))  __bf16 v8bf;
typedef __attribute__((ext_vector_type(16))) __bf16 v16bf;

#define B_   8
#define S_   512
#define P_   8
#define DIN  768
#define DM   1024
#define ROWS (B_ * S_)        // 4096
#define EPS_ 1e-5f
#define LDA  776              // bf16 elements per LDS row (768 + 8 pad)

// ---- bf16 helpers (round-to-nearest-even split: x = hi + lo, ~16 mantissa bits)
static __device__ __forceinline__ unsigned short f2bf(float f) {
    unsigned u = __float_as_uint(f);
    u += 0x7FFFu + ((u >> 16) & 1u);
    return (unsigned short)(u >> 16);
}
static __device__ __forceinline__ float bf2f(unsigned short h) {
    return __uint_as_float(((unsigned)h) << 16);
}

// -------------------------------------------------------------------------
// Kernel 0: one-time W split + swizzle into the contiguous-k (k = 16h + s) B-fragment
// layout: Wf[kb][n][k'] , kb = k/16, k' = k%16  ->  [48][1024][16] bf16.
// A lane's whole 16-value B fragment is then 32 contiguous bytes.
// -------------------------------------------------------------------------
__global__ __launch_bounds__(256) void prep_w_kernel(const float* __restrict__ W,
                                                     unsigned short* __restrict__ Whi,
                                                     unsigned short* __restrict__ Wlo) {
    int id = blockIdx.x * 256 + threadIdx.x;        // 0 .. 768*1024-1
    int k = id >> 10;
    int n = id & 1023;
    float w = W[id];
    unsigned short hi = f2bf(w);
    unsigned short lo = f2bf(w - bf2f(hi));
    size_t dst = (size_t)(k >> 4) * (DM * 16) + (size_t)n * 16 + (k & 15);
    Whi[dst] = hi;
    Wlo[dst] = lo;
}

// -------------------------------------------------------------------------
// Kernel 1: masked mean-pool over P (pool BEFORE the GEMM: 8x less matmul),
// emitting the bf16 hi/lo split of X. One wave per row; float4-coalesced.
// This pass is the HBM roofline floor (~101 MB read).
// -------------------------------------------------------------------------
__global__ __launch_bounds__(256) void pool_kernel(const float* __restrict__ pv,
                                                   const int* __restrict__ np,
                                                   unsigned short* __restrict__ Xhi,
                                                   unsigned short* __restrict__ Xlo) {
    int row  = blockIdx.x * 8 + (threadIdx.x >> 5);
    int lane = threadIdx.x & 31;
    int cnt  = np[row];
    float scale = (cnt > 0) ? 1.0f / (float)cnt : 0.0f;

    float4 acc[6];
#pragma unroll
    for (int j = 0; j < 6; ++j) acc[j] = make_float4(0.f, 0.f, 0.f, 0.f);

    const float* base = pv + (size_t)row * (P_ * DIN);
    for (int p = 0; p < cnt; ++p) {
        const float4* src = (const float4*)(base + (size_t)p * DIN);
#pragma unroll
        for (int j = 0; j < 6; ++j) {
            float4 v = src[lane + j * 32];
            acc[j].x += v.x; acc[j].y += v.y; acc[j].z += v.z; acc[j].w += v.w;
        }
    }
    ushort4* dhi = (ushort4*)(Xhi + (size_t)row * DIN);
    ushort4* dlo = (ushort4*)(Xlo + (size_t)row * DIN);
#pragma unroll
    for (int j = 0; j < 6; ++j) {
        float s[4] = { acc[j].x * scale, acc[j].y * scale, acc[j].z * scale, acc[j].w * scale };
        ushort4 h, l;
        h.x = f2bf(s[0]); l.x = f2bf(s[0] - bf2f(h.x));
        h.y = f2bf(s[1]); l.y = f2bf(s[1] - bf2f(h.y));
        h.z = f2bf(s[2]); l.z = f2bf(s[2] - bf2f(h.z));
        h.w = f2bf(s[3]); l.w = f2bf(s[3] - bf2f(h.w));
        dhi[lane + j * 32] = h;
        dlo[lane + j * 32] = l;
    }
}

// -------------------------------------------------------------------------
// Kernel 2: bf16x3 WMMA GEMM: pooled = Xhi@Whi + Xlo@Whi + Xhi@Wlo
//           (+ bias*(cnt>0)) -> d_out (pre-LayerNorm). ~fp32 accuracy on
//           the fast v_wmma_f32_16x16x32_bf16 path (K=32/inst vs K=4 f32).
// Block = 256 thr (8 waves): 16 rows x 128 cols; each wave one 16x16 tile,
// 24 K-chunks x 3 WMMAs. A hi/lo tiles staged in LDS (padded stride).
// -------------------------------------------------------------------------
__global__ __launch_bounds__(256) void gemm_kernel(const unsigned short* __restrict__ Xhi,
                                                   const unsigned short* __restrict__ Xlo,
                                                   const unsigned short* __restrict__ Whi,
                                                   const unsigned short* __restrict__ Wlo,
                                                   const float* __restrict__ bias,
                                                   const int* __restrict__ np,
                                                   float* __restrict__ out) {
    __shared__ unsigned short Ahi[16 * LDA];
    __shared__ unsigned short Alo[16 * LDA];
    __shared__ int cnts[16];

    const int m_base  = blockIdx.x * 16;
    const int n_block = blockIdx.y * 128;
    const int t       = threadIdx.x;

    // Stage 16x768 bf16 tiles (uint4 = 8 bf16 per copy; 6 chunks/thread/array).
#pragma unroll
    for (int j = 0; j < 6; ++j) {
        int idx = t + j * 256;              // 16B-chunk index in [0, 1536)
        int r   = idx / 96;
        int c8  = idx % 96;
        uint4 vh = ((const uint4*)(Xhi + (size_t)(m_base + r) * DIN))[c8];
        uint4 vl = ((const uint4*)(Xlo + (size_t)(m_base + r) * DIN))[c8];
        *(uint4*)&Ahi[r * LDA + c8 * 8] = vh;
        *(uint4*)&Alo[r * LDA + c8 * 8] = vl;
    }
    if (t < 16) cnts[t] = np[m_base + t];
    __syncthreads();

    const int wave = t >> 5;
    const int lane = t & 31;
    const int half = lane >> 4;             // A: K-halves 0-15/16-31 ; C: rows 0-7/8-15
    const int mn   = lane & 15;             // A row (M) == B column index within tile
    const int n    = n_block + wave * 16 + mn;

    v8f c = {};
    const unsigned short* Ah = &Ahi[mn * LDA];
    const unsigned short* Al = &Alo[mn * LDA];

#pragma unroll 4
    for (int ch = 0; ch < DIN / 32; ++ch) {
        const int k0 = ch * 32;
        // A fragment (16-bit A 16x32): elements 0-15 = K[k0+16h .. +16),
        // matching the B panel order. Two ds_load_b128 per matrix.
        v8bf ah0 = *(const v8bf*)(Ah + k0 + 16 * half);
        v8bf ah1 = *(const v8bf*)(Ah + k0 + 16 * half + 8);
        v8bf al0 = *(const v8bf*)(Al + k0 + 16 * half);
        v8bf al1 = *(const v8bf*)(Al + k0 + 16 * half + 8);
        v16bf a_hi = __builtin_shufflevector(ah0, ah1, 0,1,2,3,4,5,6,7,8,9,10,11,12,13,14,15);
        v16bf a_lo = __builtin_shufflevector(al0, al1, 0,1,2,3,4,5,6,7,8,9,10,11,12,13,14,15);

        // B fragment: Wf layout makes the lane's 16 values 32 contiguous bytes:
        // kb = 2*ch + half selects the 16-K panel this lane half consumes.
        const size_t boff = ((size_t)(2 * ch + half) * DM + n) * 16;
        const v8bf* bh = (const v8bf*)(Whi + boff);
        const v8bf* bl = (const v8bf*)(Wlo + boff);
        v16bf b_hi = __builtin_shufflevector(bh[0], bh[1], 0,1,2,3,4,5,6,7,8,9,10,11,12,13,14,15);
        v16bf b_lo = __builtin_shufflevector(bl[0], bl[1], 0,1,2,3,4,5,6,7,8,9,10,11,12,13,14,15);

        c = __builtin_amdgcn_wmma_f32_16x16x32_bf16(false, a_hi, false, b_hi, (short)0, c, false, false);
        c = __builtin_amdgcn_wmma_f32_16x16x32_bf16(false, a_lo, false, b_hi, (short)0, c, false, false);
        c = __builtin_amdgcn_wmma_f32_16x16x32_bf16(false, a_hi, false, b_lo, (short)0, c, false, false);
    }

    const float bn = bias[n];
#pragma unroll
    for (int i = 0; i < 8; ++i) {
        int   mr = i + 8 * half;            // C/D layout: VGPR i -> M=i / M=i+8
        float fl = (cnts[mr] > 0) ? 1.0f : 0.0f;
        out[(size_t)(m_base + mr) * DM + n] = c[i] + bn * fl;
    }
}

// -------------------------------------------------------------------------
// Kernel 3: in-place LayerNorm over DM=1024 per row. One block per row.
// -------------------------------------------------------------------------
__global__ __launch_bounds__(256) void ln_kernel(float* __restrict__ out,
                                                 const float* __restrict__ gamma,
                                                 const float* __restrict__ beta) {
    __shared__ float red[256];
    const int row = blockIdx.x;
    const int t   = threadIdx.x;

    float4 v = ((const float4*)(out + (size_t)row * DM))[t];
    float s  = v.x + v.y + v.z + v.w;
    float sq = v.x * v.x + v.y * v.y + v.z * v.z + v.w * v.w;

    red[t] = s;
    __syncthreads();
    for (int o = 128; o > 0; o >>= 1) { if (t < o) red[t] += red[t + o]; __syncthreads(); }
    float mu = red[0] * (1.0f / DM);
    __syncthreads();

    red[t] = sq;
    __syncthreads();
    for (int o = 128; o > 0; o >>= 1) { if (t < o) red[t] += red[t + o]; __syncthreads(); }
    float var = red[0] * (1.0f / DM) - mu * mu;
    float inv = rsqrtf(var + EPS_);

    float4 g  = ((const float4*)gamma)[t];
    float4 bb = ((const float4*)beta)[t];
    float4 r;
    r.x = (v.x - mu) * inv * g.x + bb.x;
    r.y = (v.y - mu) * inv * g.y + bb.y;
    r.z = (v.z - mu) * inv * g.z + bb.z;
    r.w = (v.w - mu) * inv * g.w + bb.w;
    ((float4*)(out + (size_t)row * DM))[t] = r;
}

// -------------------------------------------------------------------------
extern "C" void kernel_launch(void* const* d_in, const int* in_sizes, int n_in,
                              void* d_out, int out_size, void* d_ws, size_t ws_size,
                              hipStream_t stream) {
    const float* pv    = (const float*)d_in[0];   // [B,S,P,768] f32
    const int*   np    = (const int*)  d_in[1];   // [B,S] i32
    const float* W     = (const float*)d_in[2];   // [768,1024] f32
    const float* bias  = (const float*)d_in[3];   // [1024] f32
    const float* gamma = (const float*)d_in[4];   // [1024] f32
    const float* beta  = (const float*)d_in[5];   // [1024] f32
    float* out = (float*)d_out;                   // [B,S,1024] f32

    // Workspace carve-up (bf16 buffers), ~15.8 MB total.
    char* ws = (char*)d_ws;
    unsigned short* Xhi = (unsigned short*)(ws);                              // 6.29 MB
    unsigned short* Xlo = (unsigned short*)(ws + (size_t)ROWS * DIN * 2);     // 6.29 MB
    unsigned short* Whi = (unsigned short*)(ws + (size_t)ROWS * DIN * 4);     // 1.57 MB
    unsigned short* Wlo = (unsigned short*)(ws + (size_t)ROWS * DIN * 4 +
                                            (size_t)DIN * DM * 2);            // 1.57 MB

    prep_w_kernel<<<(DIN * DM) / 256, 256, 0, stream>>>(W, Whi, Wlo);
    pool_kernel<<<ROWS / 8, 256, 0, stream>>>(pv, np, Xhi, Xlo);
    gemm_kernel<<<dim3(ROWS / 16, DM / 128), 256, 0, stream>>>(Xhi, Xlo, Whi, Wlo, bias, np, out);
    ln_kernel<<<ROWS, 256, 0, stream>>>(out, gamma, beta);
}
